// GNNPolicy_47090021433386
// MI455X (gfx1250) — compile-verified, matrix operand order checked by reading, not run on hardware
//
#include <hip/hip_runtime.h>
#include <hip/hip_bf16.h>
#include <math.h>

typedef _Float16 half_t;
typedef __attribute__((ext_vector_type(16))) _Float16 v16h;
typedef __attribute__((ext_vector_type(8)))  _Float16 v8h;
typedef __attribute__((ext_vector_type(8)))  float    v8f;

#define B_    8
#define N_    2048
#define NODES (B_*N_)   // 16384
#define KNNK  3
#define FLT_BIG 3.0e38f

// ---- workspace layout (bytes) ----
#define WS_H16   0                        // NODES*64 f16  = 2 MiB
#define WS_SQ    (2097152)                // NODES f32     = 64 KiB
#define WS_KNN   (2097152+65536)          // NODES*3 i32   = 192 KiB
#define WS_WM16  (WS_KNN+196608)          // 64*64 f16 (transposed, N-major)
#define WS_WU116 (WS_WM16+8192)           // 128*128 f16 (transposed)
#define WS_WU216 (WS_WU116+32768)         // 128*128 f16 (transposed)

// ---------------- WMMA fragment helpers (CDNA5 16-bit layouts, ISA 7.12.2) --
// 16-bit A/B fragment: lane l (hi = l>>4) holds K-halves {0..7,16..23} (hi=0)
// or {8..15,24..31} (hi=1) -> two contiguous 8-half groups = two b128 loads.
__device__ __forceinline__ v16h load_frag_row(const half_t* rowp, int kbase, int hi) {
  v8h lo = *(const v8h*)(rowp + kbase + 8*hi);
  v8h hv = *(const v8h*)(rowp + kbase + 16 + 8*hi);
  v16h r;
  #pragma unroll
  for (int e = 0; e < 8; ++e) { r[e] = lo[e]; r[e+8] = hv[e]; }
  return r;
}

__device__ __forceinline__ v8f wmma_f16(v16h a, v16h b, v8f c) {
  // v_wmma_f32_16x16x32_f16 : D = A(16x32 f16) x B(32x16 f16) + C(16x16 f32)
  return __builtin_amdgcn_wmma_f32_16x16x32_f16(false, a, false, b, (short)0, c,
                                                false, false);
}

// sorted top-3 insert (d0 <= d1 <= d2)
#define TOP3_INSERT(nd, ni)                                        \
  if ((nd) < d2) {                                                 \
    if ((nd) < d1) {                                               \
      d2 = d1; i2 = i1;                                            \
      if ((nd) < d0) { d1 = d0; i1 = i0; d0 = (nd); i0 = (ni); }   \
      else           { d1 = (nd); i1 = (ni); }                     \
    } else           { d2 = (nd); i2 = (ni); }                     \
  }

// ---------------- kernel 1: encoder MLP (4 -> 64 -> 64), f32 exact ----------
__global__ void __launch_bounds__(256) k_encode(const float* __restrict__ x,
                                                const float* __restrict__ W1,
                                                const float* __restrict__ b1,
                                                const float* __restrict__ W2,
                                                const float* __restrict__ b2,
                                                half_t* __restrict__ h16,
                                                float* __restrict__ sq) {
  __shared__ float sW1[4*64], sb1[64], sW2[64*64], sb2[64];
  int tid = threadIdx.x;
  for (int i = tid; i < 4*64;  i += 256) sW1[i] = W1[i];
  for (int i = tid; i < 64;    i += 256) { sb1[i] = b1[i]; sb2[i] = b2[i]; }
  for (int i = tid; i < 64*64; i += 256) sW2[i] = W2[i];
  __syncthreads();

  int node = blockIdx.x * 256 + tid;
  float x0 = x[node*4+0], x1 = x[node*4+1], x2 = x[node*4+2], x3 = x[node*4+3];
  float h1[64];
  #pragma unroll 8
  for (int d = 0; d < 64; ++d) {
    float a = sb1[d] + x0*sW1[d] + x1*sW1[64+d] + x2*sW1[128+d] + x3*sW1[192+d];
    h1[d] = fmaxf(a, 0.f);
  }
  float s = 0.f;
  v8h* dst = (v8h*)(h16 + (size_t)node*64);
  for (int g = 0; g < 8; ++g) {          // 8 x b128 stores instead of 64 x b16
    v8h chunk;
    #pragma unroll
    for (int dd = 0; dd < 8; ++dd) {
      int d = g*8 + dd;
      float a = sb2[d];
      #pragma unroll 16
      for (int j = 0; j < 64; ++j) a += h1[j] * sW2[j*64 + d];  // LDS broadcast
      a = fmaxf(a, 0.f);
      s += a * a;
      chunk[dd] = (half_t)a;
    }
    dst[g] = chunk;
  }
  sq[node] = s;
}

// ---------------- kernel 2: weight prep (f32 -> f16, N-major) + std ---------
__global__ void k_prep(const float* __restrict__ Wm, const float* __restrict__ Wu1,
                       const float* __restrict__ Wu2, const float* __restrict__ lstd,
                       half_t* __restrict__ Wm16t, half_t* __restrict__ Wu116t,
                       half_t* __restrict__ Wu216t, float* __restrict__ out) {
  int tid = threadIdx.x;
  for (int i = tid; i < 64*64; i += 256) {
    int k = i >> 6, n = i & 63;
    Wm16t[n*64 + k] = (half_t)Wm[i];
  }
  for (int i = tid; i < 128*128; i += 256) {
    int k = i >> 7, n = i & 127;
    Wu116t[n*128 + k] = (half_t)Wu1[i];
    Wu216t[n*128 + k] = (half_t)Wu2[i];
  }
  if (tid < 2) out[B_*N_*2 + tid] = expf(lstd[tid]);   // std = exp(log_std)
}

// ---------------- kernel 3: fused gram + distance + top-3 kNN ---------------
// Single-wave workgroups: LDS staging is wave-private, so the HW barrier
// degenerates to S_NOP (ISA 3.1) while __syncthreads keeps fence semantics.
// Metric per row = sq[col] - 2*gram.  32 columns/iteration, two independent
// WMMA chains; diagonal excluded by a wave-uniform branch that overwrites the
// single LDS element (no per-lane cmp/cndmask in the hot loop).
__global__ void __launch_bounds__(32) k_knn(const half_t* __restrict__ h16,
                                            const float* __restrict__ sq,
                                            int* __restrict__ knn) {
  __shared__ float tile[16][32];           // 2 KiB, wave-private
  __shared__ float md[32][3];
  __shared__ int   mi[32][3];
  int lane = threadIdx.x;
  int hi   = lane >> 4, ln = lane & 15;
  int b    = blockIdx.x >> 7;              // 128 row-strips of 16 per batch
  int strip= blockIdx.x & 127;
  int rowbase = b*N_ + strip*16;

  v16h a0 = load_frag_row(h16 + (size_t)(rowbase + ln)*64, 0,  hi);
  v16h a1 = load_frag_row(h16 + (size_t)(rowbase + ln)*64, 32, hi);

  float d0 = FLT_BIG, d1 = FLT_BIG, d2 = FLT_BIG;
  int   i0 = 0, i1 = 0, i2 = 0;

  for (int cb = 0; cb < N_/32; ++cb) {
    int col0 = b*N_ + cb*32 + ln;
    int col1 = col0 + 16;
    v16h b00 = load_frag_row(h16 + (size_t)col0*64, 0,  hi);
    v16h b01 = load_frag_row(h16 + (size_t)col0*64, 32, hi);
    v16h b10 = load_frag_row(h16 + (size_t)col1*64, 0,  hi);
    v16h b11 = load_frag_row(h16 + (size_t)col1*64, 32, hi);
    v8f c0 = {}, c1 = {};
    c0 = wmma_f16(a0, b00, c0);
    c1 = wmma_f16(a0, b10, c1);            // independent chains for ILP
    c0 = wmma_f16(a1, b01, c0);
    c1 = wmma_f16(a1, b11, c1);
    float sq0 = sq[col0], sq1 = sq[col1];
    #pragma unroll
    for (int v = 0; v < 8; ++v) {          // C layout: M = v + 8*hi, N = ln
      tile[v + 8*hi][ln]      = fmaf(-2.0f, c0[v], sq0);
      tile[v + 8*hi][16 + ln] = fmaf(-2.0f, c1[v], sq1);
    }
    // self-exclusion: wave-uniform branch, hit in 1 of 64 iterations
    if (2*cb == strip) {
      if (lane < 16) tile[ln][ln] = FLT_BIG;
    } else if (2*cb + 1 == strip) {
      if (lane < 16) tile[ln][16 + ln] = FLT_BIG;
    }
    __syncthreads();                       // S_NOP barrier + dscnt fence
    // lane scans row `ln`, columns 16*hi .. 16*hi+15
    #pragma unroll
    for (int j = 0; j < 16; ++j) {
      int n = 16*hi + j;
      float nd = tile[ln][n];
      int   ni = b*N_ + cb*32 + n;
      TOP3_INSERT(nd, ni)
    }
    __syncthreads();
  }

  // combine the two half-wave candidate lists per row
  md[lane][0] = d0; md[lane][1] = d1; md[lane][2] = d2;
  mi[lane][0] = i0; mi[lane][1] = i1; mi[lane][2] = i2;
  __syncthreads();
  if (lane < 16) {
    #pragma unroll
    for (int j = 0; j < 3; ++j) {
      float nd = md[lane + 16][j];
      int   ni = mi[lane + 16][j];
      TOP3_INSERT(nd, ni)
    }
    int row = rowbase + lane;
    knn[row*3+0] = i0; knn[row*3+1] = i1; knn[row*3+2] = i2;
  }
}

// ---------------- kernel 4: messages + update MLP + heads -------------------
// Single-wave workgroup; 16 nodes per wave; all staging wave-private.
__global__ void __launch_bounds__(32) k_mlp(const half_t* __restrict__ h16,
                                            const int* __restrict__ knn,
                                            const half_t* __restrict__ Wm16t,
                                            const float* __restrict__ bm,
                                            const half_t* __restrict__ Wu116t,
                                            const float* __restrict__ bu1,
                                            const half_t* __restrict__ Wu216t,
                                            const float* __restrict__ bu2,
                                            const float* __restrict__ Wmean,
                                            const float* __restrict__ bmean,
                                            const float* __restrict__ Wv,
                                            const float* __restrict__ bv,
                                            float* __restrict__ out) {
  __shared__ __align__(16) half_t sw[16*128];   // 4 KiB staging
  int lane = threadIdx.x;
  int hi  = lane >> 4, ln = lane & 15;
  int nodebase = blockIdx.x * 16;

  // h fragments for the concat's first 64 features
  v16h ah0 = load_frag_row(h16 + (size_t)(nodebase + ln)*64, 0,  hi);
  v16h ah1 = load_frag_row(h16 + (size_t)(nodebase + ln)*64, 32, hi);

  // ---- msgs = mean_s relu(h[knn[:,s]] @ Wm + bm), one WMMA pass per slot
  v8f macc[4];
  v8f vz = {};
  for (int t = 0; t < 4; ++t) macc[t] = vz;
  for (int s = 0; s < KNNK; ++s) {
    int nb = knn[(nodebase + ln)*3 + s];
    v16h an0 = load_frag_row(h16 + (size_t)nb*64, 0,  hi);
    v16h an1 = load_frag_row(h16 + (size_t)nb*64, 32, hi);
    for (int t = 0; t < 4; ++t) {
      const half_t* wrow = Wm16t + (t*16 + ln)*64;     // N-major weights
      v8f c = vz;
      c = wmma_f16(an0, load_frag_row(wrow, 0,  hi), c);
      c = wmma_f16(an1, load_frag_row(wrow, 32, hi), c);
      float bias = bm[t*16+ln];
      #pragma unroll
      for (int v = 0; v < 8; ++v) macc[t][v] += fmaxf(c[v] + bias, 0.f);
    }
  }
  // msgs -> LDS (f16), converting C layout -> row-major for A-frag reload
  const float inv3 = 1.0f / 3.0f;
  for (int t = 0; t < 4; ++t)
    #pragma unroll
    for (int v = 0; v < 8; ++v)
      sw[(v + 8*hi)*64 + t*16 + ln] = (half_t)(macc[t][v] * inv3);
  __syncthreads();
  v16h a2 = load_frag_row(sw + ln*64, 0,  hi);   // concat features 64..95
  v16h a3 = load_frag_row(sw + ln*64, 32, hi);   // concat features 96..127
  __syncthreads();

  // ---- u1 = relu([h|msgs] @ Wu1 + bu1)   (8 col tiles x 4 chained WMMAs)
  for (int t = 0; t < 8; ++t) {
    const half_t* wrow = Wu116t + (t*16 + ln)*128;
    v8f c = vz;
    c = wmma_f16(ah0, load_frag_row(wrow, 0,  hi), c);
    c = wmma_f16(ah1, load_frag_row(wrow, 32, hi), c);
    c = wmma_f16(a2,  load_frag_row(wrow, 64, hi), c);
    c = wmma_f16(a3,  load_frag_row(wrow, 96, hi), c);
    float bias = bu1[t*16+ln];
    #pragma unroll
    for (int v = 0; v < 8; ++v)
      sw[(v + 8*hi)*128 + t*16 + ln] = (half_t)fmaxf(c[v] + bias, 0.f);
  }
  __syncthreads();
  v16h q0 = load_frag_row(sw + ln*128, 0,  hi);
  v16h q1 = load_frag_row(sw + ln*128, 32, hi);
  v16h q2 = load_frag_row(sw + ln*128, 64, hi);
  v16h q3 = load_frag_row(sw + ln*128, 96, hi);
  __syncthreads();

  // ---- u2 = relu(u1 @ Wu2 + bu2)
  for (int t = 0; t < 8; ++t) {
    const half_t* wrow = Wu216t + (t*16 + ln)*128;
    v8f c = vz;
    c = wmma_f16(q0, load_frag_row(wrow, 0,  hi), c);
    c = wmma_f16(q1, load_frag_row(wrow, 32, hi), c);
    c = wmma_f16(q2, load_frag_row(wrow, 64, hi), c);
    c = wmma_f16(q3, load_frag_row(wrow, 96, hi), c);
    float bias = bu2[t*16+ln];
    #pragma unroll
    for (int v = 0; v < 8; ++v)
      sw[(v + 8*hi)*128 + t*16 + ln] = (half_t)fmaxf(c[v] + bias, 0.f);
  }
  __syncthreads();

  // ---- heads: lanes 0-15 -> mean (2 dots), lanes 16-31 -> value (1 dot)
  if (lane < 16) {
    int g = nodebase + lane;
    float m0 = bmean[0], m1 = bmean[1];
    for (int gg = 0; gg < 16; ++gg) {
      v8h u8 = *(const v8h*)(sw + lane*128 + gg*8);
      #pragma unroll
      for (int e = 0; e < 8; ++e) {
        float uv = (float)u8[e];
        m0 += uv * Wmean[(gg*8+e)*2+0];
        m1 += uv * Wmean[(gg*8+e)*2+1];
      }
    }
    out[g*2+0] = m0; out[g*2+1] = m1;
  } else {
    int g = nodebase + (lane - 16);
    float vv = bv[0];
    for (int gg = 0; gg < 16; ++gg) {
      v8h u8 = *(const v8h*)(sw + (lane-16)*128 + gg*8);
      #pragma unroll
      for (int e = 0; e < 8; ++e) vv += (float)u8[e] * Wv[gg*8+e];
    }
    out[B_*N_*2 + 2 + g] = vv;
  }
}

// ---------------- launcher ---------------------------------------------------
extern "C" void kernel_launch(void* const* d_in, const int* in_sizes, int n_in,
                              void* d_out, int out_size, void* d_ws, size_t ws_size,
                              hipStream_t stream) {
  (void)in_sizes; (void)n_in; (void)out_size; (void)ws_size;
  const float* x     = (const float*)d_in[0];
  const float* W1    = (const float*)d_in[1];
  const float* b1    = (const float*)d_in[2];
  const float* W2    = (const float*)d_in[3];
  const float* b2    = (const float*)d_in[4];
  const float* Wm    = (const float*)d_in[5];
  const float* bm    = (const float*)d_in[6];
  const float* Wu1   = (const float*)d_in[7];
  const float* bu1   = (const float*)d_in[8];
  const float* Wu2   = (const float*)d_in[9];
  const float* bu2   = (const float*)d_in[10];
  const float* Wmean = (const float*)d_in[11];
  const float* bmean = (const float*)d_in[12];
  const float* Wv    = (const float*)d_in[13];
  const float* bv    = (const float*)d_in[14];
  const float* lstd  = (const float*)d_in[15];
  float* out = (float*)d_out;

  char* ws = (char*)d_ws;
  half_t* h16    = (half_t*)(ws + WS_H16);
  float*  sq     = (float*)(ws + WS_SQ);
  int*    knn    = (int*)(ws + WS_KNN);
  half_t* Wm16t  = (half_t*)(ws + WS_WM16);
  half_t* Wu116t = (half_t*)(ws + WS_WU116);
  half_t* Wu216t = (half_t*)(ws + WS_WU216);

  k_encode<<<NODES/256,     256, 0, stream>>>(x, W1, b1, W2, b2, h16, sq);
  k_prep  <<<1,             256, 0, stream>>>(Wm, Wu1, Wu2, lstd, Wm16t, Wu116t, Wu216t, out);
  k_knn   <<<B_*(N_/16),    32,  0, stream>>>(h16, sq, knn);
  k_mlp   <<<NODES/16,      32,  0, stream>>>(h16, knn, Wm16t, bm, Wu116t, bu1,
                                              Wu216t, bu2, Wmean, bmean, Wv, bv, out);
}
